// AMBsc_38036230373918
// MI455X (gfx1250) — compile-verified, matrix operand order checked by reading, not run on hardware
//
#include <hip/hip_runtime.h>

// Hamming similarity via +/-1 GEMM on WMMA f16:
//   matches[b,c] = (sum_d (2q-1)(2a-1) + DIM) / 2
// HBM-bound: query = 164 MB streamed once (NT loads). am is pre-converted once
// per launch into d_ws as +/-1 f16, zero-padded to [112][10240], so the main
// kernel stages B into LDS with guard-free async b128 copies.

#define DIM      10000
#define NCLS     100
#define BATCH    4096
#define NTILES   7                   // 7*16 = 112 padded columns
#define NPAD     (NTILES * 16)      // 112
#define KCHUNK   256                 // K staging chunk in LDS
#define KPAD     10240               // 40*256 >= DIM, zero padded in ws
#define NFULL    39                  // 39*256 = 9984 full chunks
#define KTAIL    9984                // tail covers K 9984..9999 (+pad zeros)
#define WAVES    4
#define BLOCK    (WAVES * 32)        // 128 threads, wave32
#define BM       (WAVES * 16)        // 64 rows per workgroup
#define WS_BYTES ((size_t)NPAD * KPAD * sizeof(_Float16))   // 2,293,760 B

typedef _Float16 v16h __attribute__((ext_vector_type(16)));
typedef _Float16 v8h  __attribute__((ext_vector_type(8)));
typedef _Float16 v4h  __attribute__((ext_vector_type(4)));
typedef float    v8f  __attribute__((ext_vector_type(8)));
typedef float    v4f  __attribute__((ext_vector_type(4)));

__device__ __forceinline__ v8h zero8h() {
    v8h r;
#pragma unroll
    for (int i = 0; i < 8; ++i) r[i] = (_Float16)0.0f;
    return r;
}

// Unguarded: 8 consecutive f32 {0,1} -> +/-1 f16. Caller guarantees k+8 <= DIM.
__device__ __forceinline__ v8h load8_pm1_nt(const float* __restrict__ p, int k) {
    v4f a = __builtin_nontemporal_load((const v4f*)(p + k));
    v4f b = __builtin_nontemporal_load((const v4f*)(p + k + 4));
    v8h r;
#pragma unroll
    for (int i = 0; i < 4; ++i) {
        r[i]     = (_Float16)(2.0f * a[i] - 1.0f);
        r[i + 4] = (_Float16)(2.0f * b[i] - 1.0f);
    }
    return r;
}

// ---------------- prep: am f32 {0,1} -> ws +/-1 f16, padded [NPAD][KPAD] ----
__global__ __launch_bounds__(256) void
am_to_pm1f16(const float* __restrict__ am, _Float16* __restrict__ ws) {
    const int idx = blockIdx.x * 256 + threadIdx.x;      // 8-half unit index
    const int col = idx / (KPAD / 8);
    const int k   = (idx % (KPAD / 8)) * 8;              // multiple of 8
    if (col >= NPAD) return;
    v8h h;
    if (col < NCLS && k < DIM) {                         // DIM % 8 == 0
        v4f a = *(const v4f*)(am + (size_t)col * DIM + k);
        v4f b = *(const v4f*)(am + (size_t)col * DIM + k + 4);
#pragma unroll
        for (int j = 0; j < 4; ++j) {
            h[j]     = (_Float16)(2.0f * a[j] - 1.0f);
            h[j + 4] = (_Float16)(2.0f * b[j] - 1.0f);
        }
    } else {
        h = zero8h();
    }
    *(v8h*)(ws + (size_t)col * KPAD + k) = h;
}

// ---- Async-stage KC halves per column from ws into LDS ---------------------
// LDS layout: [NPAD][KCHUNK] halves. One global_load_async_to_lds_b128 / unit.
// The VDST operand is the flat LDS address's low 32 bits (== LDS byte offset
// per the aperture mapping). Deriving it via ptrtoint of &bs[...] also makes
// the shared array escape, so the "memory" clobber forces the compiler to
// treat the async copies as real stores to Bs (no CSE of the B-fragment
// ds_loads afterwards).
template <int KC>
__device__ __forceinline__ void stage_async(const _Float16* __restrict__ wsB,
                                            _Float16* bs, int kbase, int tid) {
    constexpr int UNITS = NPAD * (KC / 8);               // 8-half (16 B) units
    for (int i = tid; i < UNITS; i += BLOCK) {
        const int col = i / (KC / 8);
        const int kq  = (i % (KC / 8)) * 8;
        const unsigned lds_off = (unsigned)(uintptr_t)(bs + col * KCHUNK + kq);
        const _Float16* g = wsB + (size_t)col * KPAD + kbase + kq;
        asm volatile("global_load_async_to_lds_b128 %0, %1, off"
                     :: "v"(lds_off), "v"(g) : "memory");
    }
    asm volatile("s_wait_asynccnt 0x0" ::: "memory");
}

// ---------------- main: WMMA GEMM, query streamed once ----------------------
__global__ __launch_bounds__(BLOCK) void
hd_hamming_wmma(const float* __restrict__ q,        // [BATCH, DIM] f32 {0,1}
                const _Float16* __restrict__ wsB,   // [NPAD, KPAD] +/-1 f16
                float* __restrict__ out)            // [BATCH, NCLS]
{
    __shared__ _Float16 Bs[NPAD * KCHUNK];          // 57,344 B

    const int tid  = threadIdx.x;
    const int wave = tid >> 5;
    const int lane = tid & 31;
    const int l15  = lane & 15;
    const int hsel = lane >> 4;

    const int rowBase = blockIdx.x * BM + wave * 16;
    const float* qrow = q + (size_t)(rowBase + l15) * DIM;

    v8f acc[NTILES];
#pragma unroll
    for (int t = 0; t < NTILES; ++t)
#pragma unroll
        for (int i = 0; i < 8; ++i) acc[t][i] = 0.0f;

    // ---- 39 guard-free full chunks of K=256 ----
    for (int c = 0; c < NFULL; ++c) {
        const int kbase = c * KCHUNK;
        __syncthreads();                       // prev chunk's LDS reads done
        stage_async<KCHUNK>(wsB, Bs, kbase, tid);
        __syncthreads();

        __builtin_prefetch(qrow + kbase + KCHUNK + hsel * 32, 0, 0);

#pragma unroll
        for (int s = 0; s < KCHUNK / 32; ++s) {
            const int k0 = kbase + s * 32;
            v8h alo = load8_pm1_nt(qrow, k0 + hsel * 8);
            v8h ahi = load8_pm1_nt(qrow, k0 + 16 + hsel * 8);
            v16h afrag = __builtin_shufflevector(alo, ahi,
                0, 1, 2, 3, 4, 5, 6, 7, 8, 9, 10, 11, 12, 13, 14, 15);

            const int kl = s * 32;
#pragma unroll
            for (int t = 0; t < NTILES; ++t) {
                const _Float16* bp = &Bs[(t * 16 + l15) * KCHUNK + kl];
                v8h blo = *(const v8h*)(bp + hsel * 8);
                v8h bhi = *(const v8h*)(bp + 16 + hsel * 8);
                v16h bfrag = __builtin_shufflevector(blo, bhi,
                    0, 1, 2, 3, 4, 5, 6, 7, 8, 9, 10, 11, 12, 13, 14, 15);
                acc[t] = __builtin_amdgcn_wmma_f32_16x16x32_f16(
                    false, afrag, false, bfrag, (short)0, acc[t], false, false);
            }
        }
    }

    // ---- tail: K 9984..9999 (A hi-half = 0; B padded zeros from ws) ----
    __syncthreads();
    stage_async<32>(wsB, Bs, KTAIL, tid);
    __syncthreads();
    {
        v8h alo = load8_pm1_nt(qrow, KTAIL + hsel * 8);   // all < DIM
        v8h ahi = zero8h();                               // K >= 10000
        v16h afrag = __builtin_shufflevector(alo, ahi,
            0, 1, 2, 3, 4, 5, 6, 7, 8, 9, 10, 11, 12, 13, 14, 15);
#pragma unroll
        for (int t = 0; t < NTILES; ++t) {
            const _Float16* bp = &Bs[(t * 16 + l15) * KCHUNK];
            v8h blo = *(const v8h*)(bp + hsel * 8);
            v8h bhi = *(const v8h*)(bp + 16 + hsel * 8);  // ws zeros
            v16h bfrag = __builtin_shufflevector(blo, bhi,
                0, 1, 2, 3, 4, 5, 6, 7, 8, 9, 10, 11, 12, 13, 14, 15);
            acc[t] = __builtin_amdgcn_wmma_f32_16x16x32_f16(
                false, afrag, false, bfrag, (short)0, acc[t], false, false);
        }
    }

    // ---- epilogue: matches = (dot + DIM) / 2 ----
#pragma unroll
    for (int t = 0; t < NTILES; ++t) {
        const int col = t * 16 + l15;
        if (col < NCLS) {
#pragma unroll
            for (int i = 0; i < 8; ++i) {
                const int r = rowBase + i + hsel * 8;
                out[(size_t)r * NCLS + col] = 0.5f * (acc[t][i] + (float)DIM);
            }
        }
    }
}

// ---------------- fallback (ws too small): direct staging w/ conversion -----
__global__ __launch_bounds__(BLOCK) void
hd_hamming_wmma_direct(const float* __restrict__ q,
                       const float* __restrict__ am,
                       float* __restrict__ out)
{
    __shared__ _Float16 Bs[NPAD * 128];
    const int tid  = threadIdx.x;
    const int wave = tid >> 5;
    const int lane = tid & 31;
    const int l15  = lane & 15;
    const int hsel = lane >> 4;
    const int rowBase = blockIdx.x * BM + wave * 16;
    const float* qrow = q + (size_t)(rowBase + l15) * DIM;

    v8f acc[NTILES];
#pragma unroll
    for (int t = 0; t < NTILES; ++t)
#pragma unroll
        for (int i = 0; i < 8; ++i) acc[t][i] = 0.0f;

    const int nChunks = (DIM + 127) / 128;
    for (int c = 0; c < nChunks; ++c) {
        const int kbase = c * 128;
        __syncthreads();
        for (int i = tid; i < NPAD * 32; i += BLOCK) {
            const int col = i / 32;
            const int kq  = (i % 32) * 4;
            const int k   = kbase + kq;
            v4h hv;
            if (col < NCLS && k < DIM) {
                v4f v = *(const v4f*)(am + (size_t)col * DIM + k);
#pragma unroll
                for (int j = 0; j < 4; ++j) hv[j] = (_Float16)(2.0f * v[j] - 1.0f);
            } else {
#pragma unroll
                for (int j = 0; j < 4; ++j) hv[j] = (_Float16)0.0f;
            }
            *(v4h*)&Bs[col * 128 + kq] = hv;
        }
        __syncthreads();
#pragma unroll
        for (int s = 0; s < 4; ++s) {
            const int k0 = kbase + s * 32;
            if (k0 < DIM) {
                v8h alo = (k0 + hsel * 8 + 8 <= DIM) ? load8_pm1_nt(qrow, k0 + hsel * 8) : zero8h();
                v8h ahi = (k0 + 16 + hsel * 8 + 8 <= DIM) ? load8_pm1_nt(qrow, k0 + 16 + hsel * 8) : zero8h();
                v16h afrag = __builtin_shufflevector(alo, ahi,
                    0, 1, 2, 3, 4, 5, 6, 7, 8, 9, 10, 11, 12, 13, 14, 15);
#pragma unroll
                for (int t = 0; t < NTILES; ++t) {
                    const _Float16* bp = &Bs[(t * 16 + l15) * 128 + s * 32];
                    v8h blo = *(const v8h*)(bp + hsel * 8);
                    v8h bhi = *(const v8h*)(bp + 16 + hsel * 8);
                    v16h bfrag = __builtin_shufflevector(blo, bhi,
                        0, 1, 2, 3, 4, 5, 6, 7, 8, 9, 10, 11, 12, 13, 14, 15);
                    acc[t] = __builtin_amdgcn_wmma_f32_16x16x32_f16(
                        false, afrag, false, bfrag, (short)0, acc[t], false, false);
                }
            }
        }
    }
#pragma unroll
    for (int t = 0; t < NTILES; ++t) {
        const int col = t * 16 + l15;
        if (col < NCLS) {
#pragma unroll
            for (int i = 0; i < 8; ++i) {
                const int r = rowBase + i + hsel * 8;
                out[(size_t)r * NCLS + col] = 0.5f * (acc[t][i] + (float)DIM);
            }
        }
    }
}

extern "C" void kernel_launch(void* const* d_in, const int* in_sizes, int n_in,
                              void* d_out, int out_size, void* d_ws, size_t ws_size,
                              hipStream_t stream) {
    const float* q  = (const float*)d_in[0];   // [4096, 10000] f32 {0,1}
    const float* am = (const float*)d_in[1];   // [100, 10000]  f32 {0,1}
    float* out      = (float*)d_out;           // [4096, 100]   f32

    if (ws_size >= WS_BYTES && d_ws != nullptr) {
        _Float16* wsB = (_Float16*)d_ws;
        const int prep_units  = NPAD * (KPAD / 8);           // 143,360
        const int prep_blocks = (prep_units + 255) / 256;    // 560
        am_to_pm1f16<<<prep_blocks, 256, 0, stream>>>(am, wsB);
        hd_hamming_wmma<<<BATCH / BM, BLOCK, 0, stream>>>(q, wsB, out);
    } else {
        hd_hamming_wmma_direct<<<BATCH / BM, BLOCK, 0, stream>>>(q, am, out);
    }
}
